// TransformerBlock_53214644797453
// MI455X (gfx1250) — matrix operand order repaired and checked by reading, not compiled, not run on hardware
//
#include <hip/hip_runtime.h>

// ---------------- problem constants ----------------
constexpr int kD   = 2048;   // d_model
constexpr int kH   = 16;     // heads
constexpr int kHD  = 128;    // head dim
constexpr int kFF  = 8192;   // d_ff
constexpr int kB   = 2;
constexpr int kL   = 2048;
constexpr int kM   = kB * kL;         // 4096 rows
constexpr float kLnEps = 1e-6f;

// ---------------- types ----------------
typedef __attribute__((ext_vector_type(16))) __bf16 bf16x16;
typedef __attribute__((ext_vector_type(8)))  float  v8f;
typedef __attribute__((ext_vector_type(4)))  int    v4i;

union Frag {
    uint4  u[2];
    bf16x16 v;
};

static __device__ __forceinline__ v8f wmma_bf16(bf16x16 a, bf16x16 b, v8f c) {
    return __builtin_amdgcn_wmma_f32_16x16x32_bf16(
        /*neg_a=*/false, a, /*neg_b=*/false, b,
        /*c_mod=*/(short)0, c, /*reuse_a=*/false, /*reuse_b=*/false);
}

static __device__ __forceinline__ unsigned short f2bf(float f) {
    unsigned int u = __float_as_uint(f);
    u = (u + 0x7FFFu + ((u >> 16) & 1u)) >> 16;   // round-to-nearest-even
    return (unsigned short)u;
}

static __device__ __forceinline__ float gelu_tanh(float v) {
    float u = 0.7978845608028654f * (v + 0.044715f * v * v * v);
    return 0.5f * v * (1.0f + tanhf(u));
}

// ---------------- CDNA5 async global->LDS path (probe via __has_builtin) ----------
#if defined(__AMDGCN__) && __has_builtin(__builtin_amdgcn_global_load_async_to_lds_b128) && \
    __has_builtin(__builtin_amdgcn_s_wait_asynccnt)
#define HAVE_ASYNC_LDS 1
#else
#define HAVE_ASYNC_LDS 0
#endif

typedef __attribute__((address_space(1))) v4i* gas_v4i_ptr;   // global
typedef __attribute__((address_space(3))) v4i* las_v4i_ptr;   // LDS

static __device__ __forceinline__ void async_ld_b128(const void* g, void* l) {
#if HAVE_ASYNC_LDS
    __builtin_amdgcn_global_load_async_to_lds_b128(
        (gas_v4i_ptr)(g), (las_v4i_ptr)(l), 0, 0);
#else
    *(uint4*)l = *(const uint4*)g;
#endif
}

template <int N>
static __device__ __forceinline__ void wait_async() {
#if HAVE_ASYNC_LDS
    __builtin_amdgcn_s_wait_asynccnt(N);
#endif
}

// ---------------- fp32 -> bf16 convert ----------------
__global__ void __launch_bounds__(256)
cvt_kernel(const float* __restrict__ in, unsigned short* __restrict__ out, long n) {
    long i = ((long)blockIdx.x * blockDim.x + threadIdx.x) * 4;
    long stride = (long)gridDim.x * blockDim.x * 4;
    for (; i < n; i += stride) {
        float4 f = *(const float4*)(in + i);
        out[i + 0] = f2bf(f.x);
        out[i + 1] = f2bf(f.y);
        out[i + 2] = f2bf(f.z);
        out[i + 3] = f2bf(f.w);
    }
}

// ---------------- LayerNorm (fp32 in, bf16 out), D = 2048, 256 thr/row ----------------
__global__ void __launch_bounds__(256)
ln_kernel(const float* __restrict__ x, const float* __restrict__ sc,
          const float* __restrict__ bi, unsigned short* __restrict__ out) {
    const int row = blockIdx.x;
    const int tid = threadIdx.x;
    const float* xr = x + (size_t)row * kD;
    const float4* xv = (const float4*)xr;
    float4 a = xv[tid * 2 + 0];
    float4 c = xv[tid * 2 + 1];
    float v[8] = {a.x, a.y, a.z, a.w, c.x, c.y, c.z, c.w};
    float s = 0.f, q = 0.f;
#pragma unroll
    for (int j = 0; j < 8; ++j) { s += v[j]; q += v[j] * v[j]; }
    __shared__ float rs_[256], rq_[256];
    rs_[tid] = s; rq_[tid] = q;
    __syncthreads();
    for (int off = 128; off > 0; off >>= 1) {
        if (tid < off) { rs_[tid] += rs_[tid + off]; rq_[tid] += rq_[tid + off]; }
        __syncthreads();
    }
    const float mu  = rs_[0] * (1.0f / kD);
    const float var = rq_[0] * (1.0f / kD) - mu * mu;
    const float inv = rsqrtf(var + kLnEps);
#pragma unroll
    for (int j = 0; j < 8; ++j) {
        int cc = tid * 8 + j;
        out[(size_t)row * kD + cc] = f2bf((v[j] - mu) * inv * sc[cc] + bi[cc]);
    }
}

// ---------------- generic bf16 GEMM: C[M,N] = A[M,K] @ B[K,N] (+bias,+resid,gelu) ----
// block = 256 threads (8 waves), 128x128 C tile, K step 64, double-buffered LDS,
// async global->LDS prefetch (ASYNCcnt) when available.
__global__ void __launch_bounds__(256)
gemm_bf16(const unsigned short* __restrict__ A, const unsigned short* __restrict__ Bm,
          const float* __restrict__ bias, const float* __restrict__ resid,
          float* __restrict__ outf, unsigned short* __restrict__ outb,
          int M, int N, int K, int flags /*1 = gelu*/) {
    __shared__ unsigned short As[2][128][72];    // 64 + 8 pad
    __shared__ unsigned short Bs[2][64][136];    // 128 + 8 pad

    const int tid   = threadIdx.x;
    const int lane  = tid & 31;
    const int wid   = tid >> 5;
    const int waveM = wid >> 2;      // 0..1
    const int waveN = wid & 3;       // 0..3
    const int mq    = lane & 15;
    const int hl    = lane >> 4;
    const int m0g   = blockIdx.y * 128;
    const int n0g   = blockIdx.x * 128;

    v8f acc[4][2];
#pragma unroll
    for (int mi = 0; mi < 4; ++mi)
#pragma unroll
        for (int ni = 0; ni < 2; ++ni)
            acc[mi][ni] = v8f{0.f, 0.f, 0.f, 0.f, 0.f, 0.f, 0.f, 0.f};

    auto stage = [&](int kt, int buf) {
        // A: 128 rows x 64 cols = 1024 x b128 ; B: 64 rows x 128 cols = 1024 x b128
#pragma unroll
        for (int it = 0; it < 4; ++it) {
            int i = tid + it * 256;
            int row = i >> 3, c4 = i & 7;
            async_ld_b128(A + (size_t)(m0g + row) * K + kt * 64 + c4 * 8,
                          &As[buf][row][c4 * 8]);
        }
#pragma unroll
        for (int it = 0; it < 4; ++it) {
            int i = tid + it * 256;
            int row = i >> 4, c4 = i & 15;
            async_ld_b128(Bm + (size_t)(kt * 64 + row) * N + n0g + c4 * 8,
                          &Bs[buf][row][c4 * 8]);
        }
    };

    const int ktiles = K >> 6;
    stage(0, 0);
    for (int kt = 0; kt < ktiles; ++kt) {
        const int buf = kt & 1;
        if (kt + 1 < ktiles) {
            stage(kt + 1, buf ^ 1);   // prefetch into other buffer
            wait_async<8>();          // oldest 8 (current buf) complete; 8 in flight
        } else {
            wait_async<0>();
        }
        __syncthreads();

#pragma unroll
        for (int ks = 0; ks < 2; ++ks) {
            Frag af[4], bfg[2];
#pragma unroll
            for (int mi = 0; mi < 4; ++mi) {
                int mm = waveM * 64 + mi * 16 + mq;
                af[mi].u[0] = *(const uint4*)&As[buf][mm][ks * 32 + hl * 8];
                af[mi].u[1] = *(const uint4*)&As[buf][mm][ks * 32 + 16 + hl * 8];
            }
#pragma unroll
            for (int ni = 0; ni < 2; ++ni) {
                int cb = waveN * 32 + ni * 16;
                unsigned short* bt = (unsigned short*)bfg[ni].u;
#pragma unroll
                for (int s = 0; s < 16; ++s) {
                    int kk = (s < 8) ? hl * 8 + s : 16 + hl * 8 + (s - 8);
                    bt[s] = Bs[buf][ks * 32 + kk][cb + mq];
                }
            }
#pragma unroll
            for (int mi = 0; mi < 4; ++mi)
#pragma unroll
                for (int ni = 0; ni < 2; ++ni)
                    acc[mi][ni] = wmma_bf16(af[mi].v, bfg[ni].v, acc[mi][ni]);
        }
        __syncthreads();
    }

    // epilogue
#pragma unroll
    for (int mi = 0; mi < 4; ++mi) {
#pragma unroll
        for (int ni = 0; ni < 2; ++ni) {
            int gn = n0g + waveN * 32 + ni * 16 + mq;
            float bv = bias ? bias[gn] : 0.f;
#pragma unroll
            for (int r = 0; r < 8; ++r) {
                int gm = m0g + waveM * 64 + mi * 16 + r + 8 * hl;
                float v = acc[mi][ni][r] + bv;
                if (resid) v += resid[(size_t)gm * N + gn];
                if (flags & 1) v = gelu_tanh(v);
                if (outf) outf[(size_t)gm * N + gn] = v;
                if (outb) outb[(size_t)gm * N + gn] = f2bf(v);
            }
        }
    }
}

// ---------------- flash attention (causal), bf16 q/k/v, bf16 out ----------------
// grid: (B*H, L/128). block = 256 thr = 8 waves; wave owns 16 q rows.
__global__ void __launch_bounds__(256)
attn_kernel(const unsigned short* __restrict__ Q, const unsigned short* __restrict__ Kb,
            const unsigned short* __restrict__ Vb, unsigned short* __restrict__ O) {
    __shared__ unsigned short Kts[128][40];        // K^T : [d][key]
    __shared__ unsigned short Vs[32][136];         // V   : [key][d]
    __shared__ unsigned short Pst[8][16][40];      // per-wave P tile

    const int bh = blockIdx.x;
    const int b  = bh >> 4;        // /kH
    const int h  = bh & 15;
    const int qt = blockIdx.y;

    const int tid  = threadIdx.x;
    const int lane = tid & 31;
    const int wid  = tid >> 5;
    const int mq   = lane & 15;
    const int hl   = lane >> 4;
    const float rscale = 0.08838834764831845f;     // 1/sqrt(128)

    // Q fragments: 16 rows x 128 d, directly from global (A layout)
    Frag qf[4];
    {
        const int qrow = qt * 128 + wid * 16 + mq;
        const unsigned short* qp = Q + ((size_t)(b * kL + qrow) * kH + h) * kHD;
#pragma unroll
        for (int kf = 0; kf < 4; ++kf) {
            qf[kf].u[0] = *(const uint4*)(qp + kf * 32 + hl * 8);
            qf[kf].u[1] = *(const uint4*)(qp + kf * 32 + 16 + hl * 8);
        }
    }

    v8f o[8];
#pragma unroll
    for (int dt = 0; dt < 8; ++dt) o[dt] = v8f{0.f,0.f,0.f,0.f,0.f,0.f,0.f,0.f};
    float mrun[8], lrun[8];
#pragma unroll
    for (int r = 0; r < 8; ++r) { mrun[r] = -3.0e38f; lrun[r] = 0.f; }

    const int qg = qt * 128 + wid * 16;     // wave's first q row
    const int nchunks = (qt + 1) * 4;       // causal: keys < (qt+1)*128

    for (int c = 0; c < nchunks; ++c) {
        const int k0 = c * 32;
        // ---- cooperative stage: Vs (async), Kts (manual transpose) ----
        {
            int key = tid >> 3;
            int db  = (tid & 7) * 16;
            const unsigned short* kp =
                Kb + ((size_t)(b * kL + k0 + key) * kH + h) * kHD + db;
            const unsigned short* vp =
                Vb + ((size_t)(b * kL + k0 + key) * kH + h) * kHD + db;
            async_ld_b128(vp,     &Vs[key][db]);
            async_ld_b128(vp + 8, &Vs[key][db + 8]);
            unsigned short tmp[16];
            *(uint4*)(tmp)     = *(const uint4*)(kp);
            *(uint4*)(tmp + 8) = *(const uint4*)(kp + 8);
#pragma unroll
            for (int j = 0; j < 16; ++j) Kts[db + j][key] = tmp[j];
        }
        wait_async<0>();
        __syncthreads();

        const bool skip = (k0 > qg + 15);   // wave-uniform
        if (!skip) {
            // ---- S = Q @ K^T  (16 x 32) ----
            v8f s0 = v8f{0.f,0.f,0.f,0.f,0.f,0.f,0.f,0.f};
            v8f s1 = v8f{0.f,0.f,0.f,0.f,0.f,0.f,0.f,0.f};
#pragma unroll
            for (int kf = 0; kf < 4; ++kf) {
                Frag kb0, kb1;
                unsigned short* t0 = (unsigned short*)kb0.u;
                unsigned short* t1 = (unsigned short*)kb1.u;
#pragma unroll
                for (int s = 0; s < 16; ++s) {
                    int d = kf * 32 + ((s < 8) ? hl * 8 + s : 16 + hl * 8 + (s - 8));
                    t0[s] = Kts[d][mq];
                    t1[s] = Kts[d][16 + mq];
                }
                s0 = wmma_bf16(qf[kf].v, kb0.v, s0);
                s1 = wmma_bf16(qf[kf].v, kb1.v, s1);
            }
            // ---- scale + causal mask + chunk row-max ----
            float cm[8];
#pragma unroll
            for (int r = 0; r < 8; ++r) {
                int qrow = qg + r + 8 * hl;
                float a0 = s0[r] * rscale;
                float a1 = s1[r] * rscale;
                if (k0 + mq      > qrow) a0 = -100000.0f;
                if (k0 + 16 + mq > qrow) a1 = -100000.0f;
                s0[r] = a0; s1[r] = a1;
                cm[r] = fmaxf(a0, a1);
            }
#pragma unroll
            for (int off = 1; off < 16; off <<= 1)
#pragma unroll
                for (int r = 0; r < 8; ++r)
                    cm[r] = fmaxf(cm[r], __shfl_xor(cm[r], off, 32));
            // ---- online softmax update ----
            float ps[8];
#pragma unroll
            for (int r = 0; r < 8; ++r) {
                float nm = fmaxf(mrun[r], cm[r]);
                float f  = __expf(mrun[r] - nm);
                mrun[r]  = nm;
                float p0 = __expf(s0[r] - nm);
                float p1 = __expf(s1[r] - nm);
                s0[r] = p0; s1[r] = p1;
                ps[r] = p0 + p1;
                lrun[r] *= f;
#pragma unroll
                for (int dt = 0; dt < 8; ++dt) o[dt][r] *= f;
            }
#pragma unroll
            for (int off = 1; off < 16; off <<= 1)
#pragma unroll
                for (int r = 0; r < 8; ++r)
                    ps[r] += __shfl_xor(ps[r], off, 32);
#pragma unroll
            for (int r = 0; r < 8; ++r) lrun[r] += ps[r];

            // ---- P: C-layout -> A-layout via per-wave LDS tile ----
            unsigned short* Pw = &Pst[wid][0][0];
#pragma unroll
            for (int r = 0; r < 8; ++r) {
                int row = r + 8 * hl;
                Pw[row * 40 + mq]      = f2bf(s0[r]);
                Pw[row * 40 + 16 + mq] = f2bf(s1[r]);
            }
            asm volatile("" ::: "memory");   // keep DS store->load order (same wave, in-order)
            Frag pf;
            pf.u[0] = *(const uint4*)&Pw[mq * 40 + hl * 8];
            pf.u[1] = *(const uint4*)&Pw[mq * 40 + 16 + hl * 8];
            // ---- O += P @ V ----
#pragma unroll
            for (int dt = 0; dt < 8; ++dt) {
                Frag vf;
                unsigned short* vt = (unsigned short*)vf.u;
#pragma unroll
                for (int s = 0; s < 16; ++s) {
                    int kk = (s < 8) ? hl * 8 + s : 16 + hl * 8 + (s - 8);
                    vt[s] = Vs[kk][dt * 16 + mq];
                }
                o[dt] = wmma_bf16(pf.v, vf.v, o[dt]);
            }
        }
        __syncthreads();
    }

    // ---- normalize + store bf16 ----
    float inv[8];
#pragma unroll
    for (int r = 0; r < 8; ++r) inv[r] = 1.0f / lrun[r];
#pragma unroll
    for (int dt = 0; dt < 8; ++dt)
#pragma unroll
        for (int r = 0; r < 8; ++r) {
            int qrow = qg + r + 8 * hl;
            int d    = dt * 16 + mq;
            O[((size_t)(b * kL + qrow) * kH + h) * kHD + d] = f2bf(o[dt][r] * inv[r]);
        }
}

// ---------------- host-side orchestration ----------------
extern "C" void kernel_launch(void* const* d_in, const int* in_sizes, int n_in,
                              void* d_out, int out_size, void* d_ws, size_t ws_size,
                              hipStream_t stream) {
    (void)in_sizes; (void)n_in; (void)out_size; (void)ws_size;
    const float* x    = (const float*)d_in[0];
    // d_in[1] = mask (causal, recomputed analytically)
    const float* ln1s = (const float*)d_in[2];
    const float* ln1b = (const float*)d_in[3];
    const float* wq   = (const float*)d_in[4];
    const float* wk   = (const float*)d_in[5];
    const float* wv   = (const float*)d_in[6];
    const float* wo   = (const float*)d_in[7];
    const float* ln2s = (const float*)d_in[8];
    const float* ln2b = (const float*)d_in[9];
    const float* w1   = (const float*)d_in[10];
    const float* b1   = (const float*)d_in[11];
    const float* w2   = (const float*)d_in[12];
    const float* b2   = (const float*)d_in[13];
    float* out = (float*)d_out;

    char* ws = (char*)d_ws;
    const size_t SZ_W4   = (size_t)kD * kD * 2;     // 8 MB
    const size_t SZ_WFF  = (size_t)kD * kFF * 2;    // 32 MB
    const size_t SZ_ACT  = (size_t)kM * kD * 2;     // 16 MB
    unsigned short* wq_b = (unsigned short*)(ws);
    unsigned short* wk_b = (unsigned short*)(ws + SZ_W4);
    unsigned short* wv_b = (unsigned short*)(ws + 2 * SZ_W4);
    unsigned short* wo_b = (unsigned short*)(ws + 3 * SZ_W4);
    unsigned short* w1_b = (unsigned short*)(ws + 4 * SZ_W4);
    unsigned short* w2_b = (unsigned short*)(ws + 4 * SZ_W4 + SZ_WFF);
    char* act0 = ws + 4 * SZ_W4 + 2 * SZ_WFF;
    unsigned short* h_b   = (unsigned short*)(act0);                 // ln1 out; reused for ln2 out
    unsigned short* q_b   = (unsigned short*)(act0 + SZ_ACT);
    unsigned short* k_b   = (unsigned short*)(act0 + 2 * SZ_ACT);
    unsigned short* v_b   = (unsigned short*)(act0 + 3 * SZ_ACT);
    unsigned short* ao_b  = (unsigned short*)(act0 + 4 * SZ_ACT);
    unsigned short* h1_b  = q_b;                                     // reuse q..ao (64 MB)
    float*          x2_f  = (float*)(act0 + 5 * SZ_ACT);

    // 1) weights -> bf16
    cvt_kernel<<<2048, 256, 0, stream>>>(wq, wq_b, (long)kD * kD);
    cvt_kernel<<<2048, 256, 0, stream>>>(wk, wk_b, (long)kD * kD);
    cvt_kernel<<<2048, 256, 0, stream>>>(wv, wv_b, (long)kD * kD);
    cvt_kernel<<<2048, 256, 0, stream>>>(wo, wo_b, (long)kD * kD);
    cvt_kernel<<<2048, 256, 0, stream>>>(w1, w1_b, (long)kD * kFF);
    cvt_kernel<<<2048, 256, 0, stream>>>(w2, w2_b, (long)kD * kFF);

    // 2) LN1
    ln_kernel<<<kM, 256, 0, stream>>>(x, ln1s, ln1b, h_b);

    // 3) QKV
    dim3 gq(kD / 128, kM / 128);
    gemm_bf16<<<gq, 256, 0, stream>>>(h_b, wq_b, nullptr, nullptr, nullptr, q_b, kM, kD, kD, 0);
    gemm_bf16<<<gq, 256, 0, stream>>>(h_b, wk_b, nullptr, nullptr, nullptr, k_b, kM, kD, kD, 0);
    gemm_bf16<<<gq, 256, 0, stream>>>(h_b, wv_b, nullptr, nullptr, nullptr, v_b, kM, kD, kD, 0);

    // 4) flash attention
    attn_kernel<<<dim3(kB * kH, kL / 128), 256, 0, stream>>>(q_b, k_b, v_b, ao_b);

    // 5) x2 = x + attn_out @ wo   (fp32)
    gemm_bf16<<<gq, 256, 0, stream>>>(ao_b, wo_b, nullptr, x, x2_f, nullptr, kM, kD, kD, 0);

    // 6) LN2
    ln_kernel<<<kM, 256, 0, stream>>>(x2_f, ln2s, ln2b, h_b);

    // 7) h1 = gelu(ln2h @ w1 + b1)
    gemm_bf16<<<dim3(kFF / 128, kM / 128), 256, 0, stream>>>(
        h_b, w1_b, b1, nullptr, nullptr, h1_b, kM, kFF, kD, 1);

    // 8) out = x2 + h1 @ w2 + b2  (fp32, final)
    gemm_bf16<<<gq, 256, 0, stream>>>(h1_b, w2_b, b2, x2_f, out, nullptr, kM, kD, kFF, 0);
}
